// RNNTLoss_3882650435685
// MI455X (gfx1250) — compile-verified, matrix operand order checked by reading, not run on hardware
//
#include <hip/hip_runtime.h>
#include <hip/hip_bf16.h>

typedef __attribute__((ext_vector_type(16))) _Float16 v16h;
typedef __attribute__((ext_vector_type(8)))  float    v8f;

#define NEGV (-1e30f)

__device__ __forceinline__ float logaddexpf_dev(float a, float b) {
    float m = fmaxf(a, b);
    float d = fminf(a, b) - m;
    // exp(d) in [0,1]; no inf/nan for finite inputs
    return m + __logf(1.0f + __expf(d));
}

// ---------------------------------------------------------------------------
// Phase 1: fused log-softmax reduction.
// Each wave32 handles a tile of 16 rows (a row = one (b,t,u) cell, 128 floats).
// Lane l and lane l+16 split row (l&15): 64 floats each (4x float4 loads x4).
// Row max via shfl_xor(16); sum of exp via WMMA f16 x ones (row sums are
// K-permutation invariant when B == ones, so no data shuffle into the A layout
// is required). Writes blank_lp[B,T,U1] and emit_lp[B,T,U] to workspace.
// ---------------------------------------------------------------------------
__global__ __launch_bounds__(256) void rnnt_lse_kernel(
    const float* __restrict__ acts, const int* __restrict__ labels,
    float* __restrict__ blank_out, float* __restrict__ emit_out,
    int B, int T, int U, int totalRows)
{
    const int U1   = U + 1;
    const int lane = threadIdx.x & 31;
    const int wave = threadIdx.x >> 5;
    const int tile = blockIdx.x * 8 + wave;
    const int half = lane >> 4;     // which 64-col half of the row
    const int r    = lane & 15;     // row within tile

    const long rowBase = (long)tile * 16;
    const long myRow   = rowBase + r;
    const long loadRow = (myRow < totalRows) ? myRow : 0;   // clamp, keep EXEC full for WMMA

    const float4* p = (const float4*)(acts + (size_t)loadRow * 128 + (size_t)half * 64);

    float v[64];
    #pragma unroll
    for (int i = 0; i < 16; ++i) {
        float4 q = p[i];
        v[4*i+0] = q.x; v[4*i+1] = q.y; v[4*i+2] = q.z; v[4*i+3] = q.w;
    }

    // full row max: local 64 + partner lane (xor 16)
    float m = v[0];
    #pragma unroll
    for (int i = 1; i < 64; ++i) m = fmaxf(m, v[i]);
    m = fmaxf(m, __shfl_xor(m, 16));

    // exp(x - max) packed to f16 (values in (0,1], f32 accumulate in WMMA)
    v16h a0, a1, a2, a3;
    #pragma unroll
    for (int i = 0; i < 16; ++i) a0[i] = (_Float16)__expf(v[i]      - m);
    #pragma unroll
    for (int i = 0; i < 16; ++i) a1[i] = (_Float16)__expf(v[16 + i] - m);
    #pragma unroll
    for (int i = 0; i < 16; ++i) a2[i] = (_Float16)__expf(v[32 + i] - m);
    #pragma unroll
    for (int i = 0; i < 16; ++i) a3[i] = (_Float16)__expf(v[48 + i] - m);

    v16h bones;
    #pragma unroll
    for (int i = 0; i < 16; ++i) bones[i] = (_Float16)1.0f;

    v8f c = {};
    c = __builtin_amdgcn_wmma_f32_16x16x32_f16(false, a0, false, bones, (short)0, c, false, false);
    c = __builtin_amdgcn_wmma_f32_16x16x32_f16(false, a1, false, bones, (short)0, c, false, false);
    c = __builtin_amdgcn_wmma_f32_16x16x32_f16(false, a2, false, bones, (short)0, c, false, false);
    c = __builtin_amdgcn_wmma_f32_16x16x32_f16(false, a3, false, bones, (short)0, c, false, false);

    // C layout: lane (half h, N=lane&15) holds rows M = h*8 + j in c[j].
    // Gather that row's max (lives in lane M) and finish lse = max + log(sum).
    float lse[8];
    #pragma unroll
    for (int j = 0; j < 8; ++j) {
        float rm = __shfl(m, half * 8 + j);
        lse[j]   = rm + __logf(c[j]);
    }

    // One lane per half writes its 8 rows (values replicated across N).
    if ((lane & 15) == 0) {
        #pragma unroll
        for (int j = 0; j < 8; ++j) {
            long row = rowBase + half * 8 + j;
            if (row < totalRows) {
                int b   = (int)(row / ((long)T * U1));
                int rem = (int)(row % ((long)T * U1));
                int t   = rem / U1;
                int u   = rem % U1;
                float blank = acts[(size_t)row * 128 + 0] - lse[j];   // L0/L2 hot
                blank_out[row] = blank;
                if (u < U) {
                    int lab = labels[b * U + u];
                    emit_out[((size_t)b * T + t) * U + u] =
                        acts[(size_t)row * 128 + lab] - lse[j];
                }
            }
        }
    }
}

// ---------------------------------------------------------------------------
// Phase 2: anti-diagonal DP, one wave32 per batch, 4 u-slots per lane
// (u = k*32 + lane). Both dependencies of a diagonal-d cell live on diagonal
// d-1, held in registers; the alpha[t,u-1] term is fetched with one shuffle.
// Wave-synchronous -> no barriers across the 300 diagonals.
// ---------------------------------------------------------------------------
__global__ __launch_bounds__(32) void rnnt_alpha_kernel(
    const float* __restrict__ blank_lp, const float* __restrict__ emit_lp,
    const int* __restrict__ act_lens, const int* __restrict__ label_lens,
    float* __restrict__ loss_out, int B, int T, int U)
{
    const int U1   = U + 1;
    const int b    = blockIdx.x;
    const int lane = threadIdx.x;
    const int flen = act_lens[b];
    const int glen = label_lens[b];

    const float* bl = blank_lp + (size_t)b * T * U1;
    const float* em = emit_lp  + (size_t)b * T * U;

    float prev[4];
    #pragma unroll
    for (int k = 0; k < 4; ++k) prev[k] = NEGV;

    const int nd = T + U1 - 1;
    for (int d = 0; d < nd; ++d) {
        float cur[4];
        #pragma unroll
        for (int k = 0; k < 4; ++k) {
            // alpha[t, u-1] from previous diagonal: slot (u-1) = lane-1 in
            // slot k, or lane 31 of slot k-1 when lane==0.
            float fromUp   = __shfl_up(prev[k], 1);
            float fromSlot = (k == 0) ? NEGV : __shfl(prev[k > 0 ? k - 1 : 0], 31);
            float leftPrev = (lane == 0) ? fromSlot : fromUp;

            int u = k * 32 + lane;
            int t = d - u;
            float val = NEGV;
            if (u < U1 && t >= 0 && t < T) {
                float up;
                if (t == 0) {
                    up = (u == 0) ? 0.0f : NEGV;   // init row
                } else {
                    float bv = ((t - 1) < flen) ? bl[(size_t)(t - 1) * U1 + u] : 0.0f;
                    up = prev[k] + bv;             // prev[k] = alpha[t-1,u]
                }
                float left;
                if (u == 0) {
                    left = NEGV;
                } else {
                    float ev = ((t < flen) && ((u - 1) < glen))
                                   ? em[(size_t)t * U + (u - 1)] : NEGV;
                    left = leftPrev + ev;
                }
                val = logaddexpf_dev(up, left);
                if (t == flen - 1 && u == glen) {
                    loss_out[b] = -(val + bl[(size_t)t * U1 + u]);
                }
            }
            cur[k] = val;
        }
        #pragma unroll
        for (int k = 0; k < 4; ++k) prev[k] = cur[k];
    }
}

// ---------------------------------------------------------------------------
// Phase 3: reduce the 16 per-batch losses into d_out[0].
// ---------------------------------------------------------------------------
__global__ __launch_bounds__(32) void rnnt_sum_kernel(
    const float* __restrict__ loss, float* __restrict__ out, int B)
{
    if (threadIdx.x == 0) {
        float s = 0.0f;
        for (int b = 0; b < B; ++b) s += loss[b];
        out[0] = s;
    }
}

extern "C" void kernel_launch(void* const* d_in, const int* in_sizes, int n_in,
                              void* d_out, int out_size, void* d_ws, size_t ws_size,
                              hipStream_t stream) {
    const float* acts      = (const float*)d_in[0];
    const int*   labels    = (const int*)d_in[1];
    const int*   act_lens  = (const int*)d_in[2];
    const int*   label_lens= (const int*)d_in[3];

    const int B  = in_sizes[2];                 // 16
    const int U  = in_sizes[1] / B;             // 100
    const int U1 = U + 1;                       // 101
    const int V  = 128;
    const int T  = (int)((long)in_sizes[0] / ((long)B * U1 * V));  // 200
    const int totalRows = B * T * U1;           // 323200

    float* ws        = (float*)d_ws;
    float* blank_out = ws;                                   // [B*T*U1]
    float* emit_out  = ws + (size_t)totalRows;               // [B*T*U]
    float* loss_out  = emit_out + (size_t)B * T * U;         // [B]

    const int tiles  = (totalRows + 15) / 16;   // 16 rows per wave
    const int blocks = (tiles + 7) / 8;         // 8 waves per 256-thread block

    rnnt_lse_kernel<<<blocks, 256, 0, stream>>>(acts, labels, blank_out, emit_out,
                                                B, T, U, totalRows);
    rnnt_alpha_kernel<<<B, 32, 0, stream>>>(blank_out, emit_out, act_lens, label_lens,
                                            loss_out, B, T, U);
    rnnt_sum_kernel<<<1, 32, 0, stream>>>(loss_out, (float*)d_out, B);
}
